// GraphPool_57767310131619
// MI455X (gfx1250) — matrix operand order repaired and checked by reading, not compiled, not run on hardware
//
#include <hip/hip_runtime.h>
#include <math.h>

typedef __attribute__((ext_vector_type(16))) _Float16 v16h;
typedef __attribute__((ext_vector_type(8)))  float    v8f;
typedef __attribute__((ext_vector_type(4)))  float    v4f;

constexpr int BB    = 16;       // batch
constexpr int NN    = 32768;    // rows per batch (2^15)
constexpr int FF    = 128;      // features
constexpr int KK    = NN / 2;   // top-k
constexpr int NBINS = 2048;     // 11-bit radix bins

// Monotonic sortable key for float logits (handles negatives).
__device__ __forceinline__ unsigned sortKey(float z) {
    unsigned u = __float_as_uint(z);
    return (u & 0x80000000u) ? ~u : (u | 0x80000000u);
}

// ---------------------------------------------------------------------------
// Kernel A: logits z = x·W + b via WMMA. One wave per 16-row tile.
// A = 16x32 f16 tile of x (documented A layout), B = W broadcast to all 16
// columns (each lane loads the K-range it owns per the B layout), C = f32.
// x reads use default RT policy on purpose: lines left in the 192MB L2 are
// re-hit by the gather kernel.
// ---------------------------------------------------------------------------
__global__ void scores_wmma(const float* __restrict__ x,
                            const float* __restrict__ W,
                            const float* __restrict__ bias,
                            float* __restrict__ scores)
{
    const int lane  = threadIdx.x & 31;
    const int waveB = threadIdx.x >> 5;
    const long tile = (long)blockIdx.x * (blockDim.x >> 5) + waveB;
    const long row0 = tile * 16;

    const int mrow = lane & 15;
    const int qA = (lane < 16) ? 0 : 8;   // A-matrix K group per lane half
    const int qB = (lane < 16) ? 0 : 8;   // B-matrix K group per lane half (same as A)

    // Preload W (B operand) for the 4 K-chunks of 32.
    v16h wreg[4];
#pragma unroll
    for (int c = 0; c < 4; ++c) {
        const v4f* wp = (const v4f*)(W + c * 32 + qB);
#pragma unroll
        for (int j = 0; j < 4; ++j) {
            v4f wv = wp[(j & 1) + 4 * (j >> 1)];
            wreg[c][4 * j + 0] = (_Float16)wv.x;
            wreg[c][4 * j + 1] = (_Float16)wv.y;
            wreg[c][4 * j + 2] = (_Float16)wv.z;
            wreg[c][4 * j + 3] = (_Float16)wv.w;
        }
    }

    v8f acc = {};
    const float* rp = x + (row0 + mrow) * FF;
#pragma unroll
    for (int c = 0; c < 4; ++c) {
        const int kb = 32 * c + qA;
        const v4f* ap = (const v4f*)(rp + kb);       // K qA..qA+7
        const v4f* bp = (const v4f*)(rp + kb + 16);  // K qA+16..qA+23
        v4f a0 = ap[0], a1 = ap[1], b0 = bp[0], b1 = bp[1];
        v16h ah;
        ah[0]  = (_Float16)a0.x; ah[1]  = (_Float16)a0.y;
        ah[2]  = (_Float16)a0.z; ah[3]  = (_Float16)a0.w;
        ah[4]  = (_Float16)a1.x; ah[5]  = (_Float16)a1.y;
        ah[6]  = (_Float16)a1.z; ah[7]  = (_Float16)a1.w;
        ah[8]  = (_Float16)b0.x; ah[9]  = (_Float16)b0.y;
        ah[10] = (_Float16)b0.z; ah[11] = (_Float16)b0.w;
        ah[12] = (_Float16)b1.x; ah[13] = (_Float16)b1.y;
        ah[14] = (_Float16)b1.z; ah[15] = (_Float16)b1.w;
        acc = __builtin_amdgcn_wmma_f32_16x16x32_f16(
            /*neg_a=*/false, ah, /*neg_b=*/false, wreg[c],
            /*c_mod=*/(short)0, acc, /*reuse_a=*/false, /*reuse_b=*/false);
    }

    // All B columns identical -> lane 0 holds M=0..7, lane 16 holds M=8..15.
    const float bv = bias[0];
    if (lane == 0) {
#pragma unroll
        for (int m = 0; m < 8; ++m) scores[row0 + m] = acc[m] + bv;
    } else if (lane == 16) {
#pragma unroll
        for (int m = 0; m < 8; ++m) scores[row0 + 8 + m] = acc[m] + bv;
    }
}

// ---------------------------------------------------------------------------
// Kernel B: per-batch radix select (one block per batch).
// Pass 1: 2048-bin histogram of key[31:21] -> descending exclusive prefixes
//         (deterministic output base per bin) + boundary bin containing rank k.
// Pass 2: refine boundary bin with key[20:10] -> exact-to-22-bit threshold.
// Score reads are 16B-vectorized (2MB buffer, L2-resident).
// ---------------------------------------------------------------------------
__global__ void select_kernel(const float* __restrict__ scores,
                              unsigned* __restrict__ binStart,
                              unsigned* __restrict__ meta,
                              unsigned* __restrict__ ctrs)
{
    __shared__ unsigned hist[NBINS];
    __shared__ unsigned sThrBin, sBStart;
    const int b   = blockIdx.x;
    const int tid = threadIdx.x;
    const v4f* sc4 = (const v4f*)(scores + (long)b * NN);

    for (int i = tid; i < NBINS; i += blockDim.x) hist[i] = 0;
    __syncthreads();
    for (int i = tid; i < NN / 4; i += blockDim.x) {
        v4f v = sc4[i];
        atomicAdd(&hist[sortKey(v.x) >> 21], 1u);
        atomicAdd(&hist[sortKey(v.y) >> 21], 1u);
        atomicAdd(&hist[sortKey(v.z) >> 21], 1u);
        atomicAdd(&hist[sortKey(v.w) >> 21], 1u);
    }
    __syncthreads();
    if (tid == 0) {
        unsigned running = 0, thrBin = 0, bstart = 0;
        for (int bin = NBINS - 1; bin >= 0; --bin) {
            unsigned cnt = hist[bin];
            hist[bin] = running;  // count of keys in strictly higher bins
            if (running < (unsigned)KK && running + cnt >= (unsigned)KK) {
                thrBin = (unsigned)bin; bstart = running;
            }
            running += cnt;
        }
        sThrBin = thrBin; sBStart = bstart;
    }
    __syncthreads();
    const unsigned thrBin = sThrBin, bstart = sBStart;
    for (int i = tid; i < NBINS; i += blockDim.x)
        binStart[b * NBINS + i] = hist[i];
    __syncthreads();

    // Pass 2: sub-histogram of the boundary bin (bits 20:10).
    for (int i = tid; i < NBINS; i += blockDim.x) hist[i] = 0;
    __syncthreads();
    for (int i = tid; i < NN / 4; i += blockDim.x) {
        v4f v = sc4[i];
        unsigned k0 = sortKey(v.x), k1 = sortKey(v.y);
        unsigned k2 = sortKey(v.z), k3 = sortKey(v.w);
        if ((k0 >> 21) == thrBin) atomicAdd(&hist[(k0 >> 10) & (NBINS - 1)], 1u);
        if ((k1 >> 21) == thrBin) atomicAdd(&hist[(k1 >> 10) & (NBINS - 1)], 1u);
        if ((k2 >> 21) == thrBin) atomicAdd(&hist[(k2 >> 10) & (NBINS - 1)], 1u);
        if ((k3 >> 21) == thrBin) atomicAdd(&hist[(k3 >> 10) & (NBINS - 1)], 1u);
    }
    __syncthreads();
    if (tid == 0) {
        unsigned running = 0, thrSub = 0, high = 0;
        for (int bin = NBINS - 1; bin >= 0; --bin) {
            unsigned cnt = hist[bin];
            if (bstart + running < (unsigned)KK &&
                bstart + running + cnt >= (unsigned)KK) {
                thrSub = (unsigned)bin; high = running; break;
            }
            running += cnt;
        }
        unsigned* m = meta + b * 8;
        m[0] = thrBin;
        m[1] = thrSub;
        m[2] = bstart;                          // base for sub > thrSub
        m[3] = bstart + high;                   // base for ties (sub == thrSub)
        m[4] = (unsigned)KK - (bstart + high);  // tie quota
        ctrs[2 * b + 0] = 0;                    // high-sub placement counter
        ctrs[2 * b + 1] = 0;                    // tie counter
    }
}

// ---------------------------------------------------------------------------
// Kernel C: compaction + gather + scale. One wave per row; lane 0 decides
// accept/position, all 32 lanes move the 512-byte row scaled by sigmoid(z).
// x loads use RT (hit L2 lines left by kernel A); output stores are
// non-temporal (write-once data must not evict x from the 192MB L2).
// ---------------------------------------------------------------------------
__global__ void gather_kernel(const float* __restrict__ x,
                              const float* __restrict__ scores,
                              unsigned* __restrict__ binStart,
                              const unsigned* __restrict__ meta,
                              unsigned* __restrict__ ctrs,
                              float* __restrict__ out,
                              int* __restrict__ idsOut)
{
    const int lane = threadIdx.x & 31;
    const int wave = threadIdx.x >> 5;
    const long r = (long)blockIdx.x * 8 + wave;  // flattened row index
    const int b = (int)(r >> 15);                // NN == 2^15
    const int n = (int)(r & (NN - 1));

    int pos = -1;
    float z = 0.0f;
    if (lane == 0) {
        z = scores[r];
        unsigned key = sortKey(z);
        unsigned bin = key >> 21;
        const unsigned* m = meta + b * 8;
        if (bin > m[0]) {
            pos = (int)atomicAdd(&binStart[b * NBINS + bin], 1u);
        } else if (bin == m[0]) {
            unsigned sub = (key >> 10) & (NBINS - 1);
            if (sub > m[1]) {
                pos = (int)(m[2] + atomicAdd(&ctrs[2 * b + 0], 1u));
            } else if (sub == m[1]) {
                unsigned t = atomicAdd(&ctrs[2 * b + 1], 1u);
                if (t < m[4]) pos = (int)(m[3] + t);
            }
        }
    }
    pos = __shfl(pos, 0, 32);
    z   = __shfl(z, 0, 32);

    if (pos >= 0) {
        const float val = 1.0f / (1.0f + __expf(-z));  // sigmoid(logit)
        v4f v = *((const v4f*)(x + r * FF) + lane);
        v.x *= val; v.y *= val; v.z *= val; v.w *= val;
        __builtin_nontemporal_store(
            v, (v4f*)(out + ((long)b * KK + pos) * FF) + lane);
        if (lane == 0)
            __builtin_nontemporal_store(n, idsOut + (long)b * KK + pos);
    }
}

// ---------------------------------------------------------------------------
extern "C" void kernel_launch(void* const* d_in, const int* in_sizes, int n_in,
                              void* d_out, int out_size, void* d_ws, size_t ws_size,
                              hipStream_t stream)
{
    (void)in_sizes; (void)n_in; (void)out_size; (void)ws_size;
    const float* x    = (const float*)d_in[0];
    const float* W    = (const float*)d_in[1];
    const float* bias = (const float*)d_in[2];

    // Workspace layout (all 4-byte units):
    float*    scoresBuf = (float*)d_ws;                          // BB*NN logits
    unsigned* binStart  = (unsigned*)d_ws + (size_t)BB * NN;     // BB*NBINS
    unsigned* meta      = binStart + (size_t)BB * NBINS;         // BB*8
    unsigned* ctrs      = meta + (size_t)BB * 8;                 // BB*2

    float* out    = (float*)d_out;                               // [B,K,F]
    int*   idsOut = (int*)d_out + (size_t)BB * KK * FF;          // [B,K] ids

    // A: 32768 tiles of 16 rows, 8 waves/block -> 4096 blocks.
    scores_wmma<<<(BB * NN / 16) / 8, 256, 0, stream>>>(x, W, bias, scoresBuf);
    // B: one block per batch.
    select_kernel<<<BB, 1024, 0, stream>>>(scoresBuf, binStart, meta, ctrs);
    // C: one wave per row -> 524288 / 8 = 65536 blocks.
    gather_kernel<<<(BB * NN) / 8, 256, 0, stream>>>(x, scoresBuf, binStart,
                                                     meta, ctrs, out, idsOut);
}
